// dmpnn_11338713662118
// MI455X (gfx1250) — matrix operand order repaired and checked by reading, not compiled, not run on hardware
//
#include <hip/hip_runtime.h>
#include <math.h>

// ---------------- problem constants (match reference) ----------------
#define NNODES   100000
#define EHALF    800000
#define NEDGES   (2*EHALF)      // 1,600,000
#define BGRAPH   256
#define NFEAT    64
#define EFEAT    16
#define EOUTC    64
#define NOUTC    128
#define EXTRAF   10
#define KEXP     8
#define HID      256
#define NROUNDS  3
#define MOE_IN   (NOUTC + EXTRAF)   // 138
#define KPAD_EDGE 96                // 80 padded to 3 chunks of 32

typedef _Float16 v16h __attribute__((ext_vector_type(16)));
typedef _Float16 v8h  __attribute__((ext_vector_type(8)));
typedef float    v8f  __attribute__((ext_vector_type(8)));

__device__ __forceinline__ float prelu_f(float x, float a) {
    return x > 0.f ? x : a * x;
}

// ---------------- small utility kernels ----------------
__global__ void zero_f32(float* __restrict__ p, int n) {
    int i = blockIdx.x * blockDim.x + threadIdx.x;
    if (i < n) p[i] = 0.f;
}

__global__ void build_edges(const int* __restrict__ sh, const int* __restrict__ dh,
                            int* __restrict__ src, int* __restrict__ dst, int eh) {
    int i = blockIdx.x * blockDim.x + threadIdx.x;
    if (i >= eh) return;
    int s = sh[i], d = dh[i];
    src[2*i]   = s; src[2*i+1] = d;
    dst[2*i]   = d; dst[2*i+1] = s;
}

// WT[n][k] = (k < K) ? W[k*Nout + n] : 0   (f32 -> f16, transpose, K padded)
__global__ void transpose_w_f16(const float* __restrict__ W, _Float16* __restrict__ WT,
                                int K, int Kpad, int Nout) {
    int idx = blockIdx.x * blockDim.x + threadIdx.x;
    int total = Nout * Kpad;
    if (idx >= total) return;
    int n = idx / Kpad, k = idx - n * Kpad;
    float v = (k < K) ? W[(size_t)k * Nout + n] : 0.f;
    WT[(size_t)n * Kpad + k] = (_Float16)v;
}

// sum0[dst[e]][c] += h[e][c]   (f16 edge state -> f32 accumulator)
__global__ void scatter_edge_sum(const _Float16* __restrict__ h, const int* __restrict__ dst,
                                 float* __restrict__ sum0) {
    long long idx = (long long)blockIdx.x * blockDim.x + threadIdx.x;
    if (idx >= (long long)NEDGES * EOUTC) return;
    int e = (int)(idx >> 6);
    int c = (int)(idx & 63);
    atomicAdd(&sum0[(size_t)dst[e] * EOUTC + c], (float)h[idx]);
}

// solute[node_graph[n]][c] += node_h[n][c]
__global__ void scatter_readout(const float* __restrict__ node_h, const int* __restrict__ ng,
                                float* __restrict__ solute) {
    long long idx = (long long)blockIdx.x * blockDim.x + threadIdx.x;
    if (idx >= (long long)NNODES * NOUTC) return;
    int n = (int)(idx >> 7);
    int c = (int)(idx & 127);
    atomicAdd(&solute[(size_t)ng[n] * NOUTC + c], node_h[idx]);
}

__global__ void build_x(const float* __restrict__ solute, const float* __restrict__ extra,
                        float* __restrict__ x) {
    int idx = blockIdx.x * blockDim.x + threadIdx.x;
    if (idx >= BGRAPH * MOE_IN) return;
    int b = idx / MOE_IN, c = idx - b * MOE_IN;
    x[idx] = (c < NOUTC) ? solute[(size_t)b * NOUTC + c] : extra[(size_t)b * EXTRAF + (c - NOUTC)];
}

// ---------------- WMMA fragment helpers (CDNA5 wave32 layouts) ----------------
// A (16x32 f16): lane -> M = lane&15; elems 0..7  = K 8*(lane>>4)+0..7,
//                                   elems 8..15 = K 16+8*(lane>>4)+0..7
// B (32x16 f16): lane -> N = lane&15; elems 0..7  = K 8*(lane>>4)+0..7,
//                elems 8..15 = K 16+8*(lane>>4)+0..7 (same as A; two runs of W^T[N][K])
// C/D (16x16 f32): lane -> N = lane&15; vgpr r -> M = r + 8*(lane>>4)
union FragH { v16h v; v8h p[2]; };

// ---------------- edge embedding: h0 = PReLU([nf[src], ef] @ W_edge) ----------------
__global__ void __launch_bounds__(256) edge_init_wmma(
    const float* __restrict__ node_feat, const float* __restrict__ edge_feat,
    const int* __restrict__ src, const _Float16* __restrict__ WT,  // [64][96]
    const float* __restrict__ a1p,
    _Float16* __restrict__ h0, _Float16* __restrict__ h)
{
    __shared__ alignas(16) _Float16 sW[64 * KPAD_EDGE];       // 12 KB
    __shared__ alignas(16) _Float16 sA[8][16 * KPAD_EDGE];    // 24 KB
    const int tid = threadIdx.x, wave = tid >> 5, lane = tid & 31;
    for (int i = tid; i < 64 * KPAD_EDGE; i += 256) sW[i] = WT[i];

    const int tile  = blockIdx.x * 8 + wave;
    const int ebase = tile * 16;
    {   // stage A tile: row = lane>>1, cols (lane&1)*48..+47
        const int row = lane >> 1;
        const int e   = ebase + row;
        const int s   = src[e];
        const int c0  = (lane & 1) * 48;
        const float* nf = node_feat + (size_t)s * NFEAT;
        const float* ef = edge_feat + (size_t)e * EFEAT;
        _Float16* dA = &sA[wave][row * KPAD_EDGE];
        #pragma unroll
        for (int j = 0; j < 48; ++j) {
            int c = c0 + j;
            float v = (c < NFEAT) ? nf[c] : ((c < NFEAT + EFEAT) ? ef[c - NFEAT] : 0.f);
            dA[c] = (_Float16)v;
        }
    }
    __syncthreads();

    const int half = lane >> 4, mr = lane & 15;
    const int kb8 = half * 8, kb16 = half * 16, mrow0 = half * 8;

    v8f acc[4];
    #pragma unroll
    for (int nt = 0; nt < 4; ++nt)
        #pragma unroll
        for (int r = 0; r < 8; ++r) acc[nt][r] = 0.f;

    const _Float16* arow = &sA[wave][mr * KPAD_EDGE];
    #pragma unroll
    for (int kc = 0; kc < 3; ++kc) {
        FragH a;
        a.p[0] = *(const v8h*)(arow + kc * 32 + kb8);
        a.p[1] = *(const v8h*)(arow + kc * 32 + 16 + kb8);
        #pragma unroll
        for (int nt = 0; nt < 4; ++nt) {
            FragH b;
            const _Float16* wr = &sW[(nt * 16 + mr) * KPAD_EDGE + kc * 32 + kb8];
            b.p[0] = *(const v8h*)(wr);
            b.p[1] = *(const v8h*)(wr + 16);
            acc[nt] = __builtin_amdgcn_wmma_f32_16x16x32_f16(
                false, a.v, false, b.v, (short)0, acc[nt], false, false);
        }
    }

    const float a1 = a1p[0];
    #pragma unroll
    for (int nt = 0; nt < 4; ++nt)
        #pragma unroll
        for (int r = 0; r < 8; ++r) {
            _Float16 hv = (_Float16)prelu_f(acc[nt][r], a1);
            size_t idx = (size_t)(ebase + mrow0 + r) * EOUTC + nt * 16 + mr;
            h0[idx] = hv;
            h[idx]  = hv;
        }
}

// ---------------- message round: h = PReLU((sum0[src]-h[rev]) @ W_upd + h0) ----------------
__global__ void __launch_bounds__(256) edge_update_wmma(
    const float* __restrict__ sum0, const int* __restrict__ src,
    const _Float16* __restrict__ WT,  // [64][64]
    const float* __restrict__ a1p,
    const _Float16* __restrict__ h0, _Float16* __restrict__ h)
{
    __shared__ alignas(16) _Float16 sW[64 * 64];       // 8 KB
    __shared__ alignas(16) _Float16 sA[8][16 * 64];    // 16 KB
    const int tid = threadIdx.x, wave = tid >> 5, lane = tid & 31;
    for (int i = tid; i < 64 * 64; i += 256) sW[i] = WT[i];

    const int tile  = blockIdx.x * 8 + wave;
    const int ebase = tile * 16;
    {   // m = sum0[src[e]] - h[e^1]  (rev stays inside the 16-aligned tile)
        const int row = lane >> 1;
        const int e   = ebase + row;
        const int s   = src[e];
        const int er  = e ^ 1;
        const int c0  = (lane & 1) * 32;
        const float*    srow = sum0 + (size_t)s * EOUTC;
        const _Float16* hr   = h + (size_t)er * EOUTC;
        _Float16* dA = &sA[wave][row * EOUTC];
        #pragma unroll
        for (int j = 0; j < 32; ++j) {
            int c = c0 + j;
            dA[c] = (_Float16)(srow[c] - (float)hr[c]);
        }
    }
    __syncthreads();

    const int half = lane >> 4, mr = lane & 15;
    const int kb8 = half * 8, kb16 = half * 16, mrow0 = half * 8;

    v8f acc[4];   // seed accumulator with h0 residual
    #pragma unroll
    for (int nt = 0; nt < 4; ++nt)
        #pragma unroll
        for (int r = 0; r < 8; ++r)
            acc[nt][r] = (float)h0[(size_t)(ebase + mrow0 + r) * EOUTC + nt * 16 + mr];

    const _Float16* arow = &sA[wave][mr * EOUTC];
    #pragma unroll
    for (int kc = 0; kc < 2; ++kc) {
        FragH a;
        a.p[0] = *(const v8h*)(arow + kc * 32 + kb8);
        a.p[1] = *(const v8h*)(arow + kc * 32 + 16 + kb8);
        #pragma unroll
        for (int nt = 0; nt < 4; ++nt) {
            FragH b;
            const _Float16* wr = &sW[(nt * 16 + mr) * EOUTC + kc * 32 + kb8];
            b.p[0] = *(const v8h*)(wr);
            b.p[1] = *(const v8h*)(wr + 16);
            acc[nt] = __builtin_amdgcn_wmma_f32_16x16x32_f16(
                false, a.v, false, b.v, (short)0, acc[nt], false, false);
        }
    }

    const float a1 = a1p[0];
    #pragma unroll
    for (int nt = 0; nt < 4; ++nt)
        #pragma unroll
        for (int r = 0; r < 8; ++r)
            h[(size_t)(ebase + mrow0 + r) * EOUTC + nt * 16 + mr] =
                (_Float16)prelu_f(acc[nt][r], a1);
}

// ---------------- node layer: node_h = PReLU([nf, m_node] @ W_node + b_node) ----------------
#define NTILES_NODE (NNODES / 16)   // 6250
__global__ void __launch_bounds__(128) node_out_wmma(
    const float* __restrict__ node_feat, const float* __restrict__ sum0,
    const _Float16* __restrict__ WT,   // [128][128]
    const float* __restrict__ b_node, const float* __restrict__ a1p,
    float* __restrict__ node_h)
{
    __shared__ alignas(16) _Float16 sW[128 * 128];     // 32 KB
    __shared__ alignas(16) _Float16 sA[4][16 * 128];   // 16 KB
    const int tid = threadIdx.x, wave = tid >> 5, lane = tid & 31;
    for (int i = tid; i < 128 * 128; i += 128) sW[i] = WT[i];

    const int tile = blockIdx.x * 4 + wave;
    const bool active = tile < NTILES_NODE;
    const int nbase = tile * 16;
    if (active) {
        const int row = lane >> 1;
        const int n   = nbase + row;
        const int c0  = (lane & 1) * 64;
        const float* nf = node_feat + (size_t)n * NFEAT;
        const float* ms = sum0 + (size_t)n * EOUTC;
        _Float16* dA = &sA[wave][row * 128];
        #pragma unroll
        for (int j = 0; j < 64; ++j) {
            int c = c0 + j;
            dA[c] = (_Float16)((c < NFEAT) ? nf[c] : ms[c - NFEAT]);
        }
    }
    __syncthreads();
    if (!active) return;

    const int half = lane >> 4, mr = lane & 15;
    const int kb8 = half * 8, kb16 = half * 16, mrow0 = half * 8;

    v8f acc[8];   // seed with bias
    #pragma unroll
    for (int nt = 0; nt < 8; ++nt) {
        float bb = b_node[nt * 16 + mr];
        #pragma unroll
        for (int r = 0; r < 8; ++r) acc[nt][r] = bb;
    }

    const _Float16* arow = &sA[wave][mr * 128];
    #pragma unroll
    for (int kc = 0; kc < 4; ++kc) {
        FragH a;
        a.p[0] = *(const v8h*)(arow + kc * 32 + kb8);
        a.p[1] = *(const v8h*)(arow + kc * 32 + 16 + kb8);
        #pragma unroll
        for (int nt = 0; nt < 8; ++nt) {
            FragH b;
            const _Float16* wr = &sW[(nt * 16 + mr) * 128 + kc * 32 + kb8];
            b.p[0] = *(const v8h*)(wr);
            b.p[1] = *(const v8h*)(wr + 16);
            acc[nt] = __builtin_amdgcn_wmma_f32_16x16x32_f16(
                false, a.v, false, b.v, (short)0, acc[nt], false, false);
        }
    }

    const float a1 = a1p[0];
    #pragma unroll
    for (int nt = 0; nt < 8; ++nt)
        #pragma unroll
        for (int r = 0; r < 8; ++r)
            node_h[(size_t)(nbase + mrow0 + r) * NOUTC + nt * 16 + mr] =
                prelu_f(acc[nt][r], a1);
}

// ---------------- MoE head (tiny: B=256) ----------------
__global__ void __launch_bounds__(256) moe_expert(
    const float* __restrict__ x,
    const float* __restrict__ We1, const float* __restrict__ be1,
    const float* __restrict__ We2, const float* __restrict__ be2,
    const float* __restrict__ We3, const float* __restrict__ be3,
    const float* __restrict__ a_moe_p, float* __restrict__ eo)
{
    const int b = blockIdx.x >> 3, k = blockIdx.x & 7, t = threadIdx.x;
    __shared__ float sx[MOE_IN + 6], s1[HID], s2[HID], sred[HID];
    const float am = a_moe_p[0];
    if (t < MOE_IN) sx[t] = x[(size_t)b * MOE_IN + t];
    __syncthreads();

    float acc = be1[(size_t)k * HID + t];
    const float* w1 = We1 + (size_t)k * MOE_IN * HID + t;
    for (int i = 0; i < MOE_IN; ++i) acc += sx[i] * w1[(size_t)i * HID];
    s1[t] = prelu_f(acc, am);
    __syncthreads();

    acc = be2[(size_t)k * HID + t];
    const float* w2 = We2 + (size_t)k * HID * HID + t;
    for (int i = 0; i < HID; ++i) acc += s1[i] * w2[(size_t)i * HID];
    s2[t] = prelu_f(acc, am);
    __syncthreads();

    sred[t] = s2[t] * We3[(size_t)k * HID + t];
    __syncthreads();
    for (int s = HID / 2; s > 0; s >>= 1) {
        if (t < s) sred[t] += sred[t + s];
        __syncthreads();
    }
    if (t == 0) eo[b * KEXP + k] = sred[0] + be3[k];
}

__global__ void __launch_bounds__(256) moe_gate(
    const float* __restrict__ x,
    const float* __restrict__ Wg1, const float* __restrict__ bg1,
    const float* __restrict__ Wg2, const float* __restrict__ bg2,
    const float* __restrict__ Wg3, const float* __restrict__ bg3,
    const float* __restrict__ a_moe_p, const float* __restrict__ eo,
    const float* __restrict__ W_f, const float* __restrict__ b_f,
    float* __restrict__ out)
{
    const int b = blockIdx.x, t = threadIdx.x;
    __shared__ float sx[MOE_IN + 6], s1[HID], s2[HID], sz[KEXP];
    const float am = a_moe_p[0];
    if (t < MOE_IN) sx[t] = x[(size_t)b * MOE_IN + t];
    __syncthreads();

    float acc = bg1[t];
    for (int i = 0; i < MOE_IN; ++i) acc += sx[i] * Wg1[(size_t)i * HID + t];
    s1[t] = prelu_f(acc, am);
    __syncthreads();

    acc = bg2[t];
    for (int i = 0; i < HID; ++i) acc += s1[i] * Wg2[(size_t)i * HID + t];
    s2[t] = prelu_f(acc, am);
    __syncthreads();

    if (t < KEXP) {
        float z = bg3[t];
        for (int i = 0; i < HID; ++i) z += s2[i] * Wg3[(size_t)i * KEXP + t];
        sz[t] = z;
    }
    __syncthreads();

    if (t == 0) {
        float mx = sz[0];
        for (int k = 1; k < KEXP; ++k) mx = fmaxf(mx, sz[k]);
        float ex[KEXP], den = 0.f;
        for (int k = 0; k < KEXP; ++k) { ex[k] = expf(sz[k] - mx); den += ex[k]; }
        float moe = 0.f;
        for (int k = 0; k < KEXP; ++k) moe += eo[b * KEXP + k] * (ex[k] / den);
        out[b] = moe * W_f[0] + b_f[0];
    }
}

// ---------------- host wiring ----------------
extern "C" void kernel_launch(void* const* d_in, const int* in_sizes, int n_in,
                              void* d_out, int out_size, void* d_ws, size_t ws_size,
                              hipStream_t stream)
{
    (void)in_sizes; (void)n_in; (void)out_size; (void)ws_size;
    const float* node_feat  = (const float*)d_in[0];
    const float* edge_feat  = (const float*)d_in[1];
    const float* extra      = (const float*)d_in[2];
    const int*   src_half   = (const int*)d_in[3];
    const int*   dst_half   = (const int*)d_in[4];
    const int*   node_graph = (const int*)d_in[5];
    const float* W_edge     = (const float*)d_in[6];
    const float* a1         = (const float*)d_in[7];
    const float* W_upd      = (const float*)d_in[8];
    const float* W_node     = (const float*)d_in[9];
    const float* b_node     = (const float*)d_in[10];
    const float* We1 = (const float*)d_in[11]; const float* be1 = (const float*)d_in[12];
    const float* We2 = (const float*)d_in[13]; const float* be2 = (const float*)d_in[14];
    const float* We3 = (const float*)d_in[15]; const float* be3 = (const float*)d_in[16];
    const float* Wg1 = (const float*)d_in[17]; const float* bg1 = (const float*)d_in[18];
    const float* Wg2 = (const float*)d_in[19]; const float* bg2 = (const float*)d_in[20];
    const float* Wg3 = (const float*)d_in[21]; const float* bg3 = (const float*)d_in[22];
    const float* a_moe = (const float*)d_in[23];
    const float* W_f   = (const float*)d_in[24];
    const float* b_f   = (const float*)d_in[25];
    float* out = (float*)d_out;

    // workspace carve-out (~500 MB total; f16 edge state halves HBM traffic)
    char* ws = (char*)d_ws;
    size_t off = 0;
    auto carve = [&](size_t bytes) -> void* {
        void* p = ws + off;
        off = (off + bytes + 255) & ~(size_t)255;
        return p;
    };
    int*       srcE    = (int*)      carve((size_t)NEDGES * 4);
    int*       dstE    = (int*)      carve((size_t)NEDGES * 4);
    _Float16*  h0      = (_Float16*) carve((size_t)NEDGES * EOUTC * 2);
    _Float16*  h       = (_Float16*) carve((size_t)NEDGES * EOUTC * 2);
    float*     sum0    = (float*)    carve((size_t)NNODES * EOUTC * 4);
    float*     node_h  = (float*)    carve((size_t)NNODES * NOUTC * 4);
    _Float16*  WTedge  = (_Float16*) carve((size_t)64 * KPAD_EDGE * 2);
    _Float16*  WTupd   = (_Float16*) carve((size_t)64 * 64 * 2);
    _Float16*  WTnode  = (_Float16*) carve((size_t)128 * 128 * 2);
    float*     solute  = (float*)    carve((size_t)BGRAPH * NOUTC * 4);
    float*     xmoe    = (float*)    carve((size_t)BGRAPH * MOE_IN * 4);
    float*     eo      = (float*)    carve((size_t)BGRAPH * KEXP * 4);

    const int T = 256;

    // 0) edge index expansion + f16 transposed weights
    build_edges<<<(EHALF + T - 1) / T, T, 0, stream>>>(src_half, dst_half, srcE, dstE, EHALF);
    transpose_w_f16<<<(64 * KPAD_EDGE + T - 1) / T, T, 0, stream>>>(W_edge, WTedge, NFEAT + EFEAT, KPAD_EDGE, EOUTC);
    transpose_w_f16<<<(64 * 64 + T - 1) / T, T, 0, stream>>>(W_upd, WTupd, EOUTC, 64, EOUTC);
    transpose_w_f16<<<(128 * 128 + T - 1) / T, T, 0, stream>>>(W_node, WTnode, NFEAT + EOUTC, 128, NOUTC);

    // 1) edge embedding (WMMA, 16-edge tiles, 8 waves/block)
    edge_init_wmma<<<NEDGES / 128, T, 0, stream>>>(node_feat, edge_feat, srcE, WTedge, a1, h0, h);

    // 2) message-passing rounds: segment-sum -> WMMA update (in place; rev stays in-tile)
    const int nSum = NNODES * EOUTC;
    const long long nScat = (long long)NEDGES * EOUTC;
    for (int r = 0; r < NROUNDS; ++r) {
        zero_f32<<<(nSum + T - 1) / T, T, 0, stream>>>(sum0, nSum);
        scatter_edge_sum<<<(unsigned)((nScat + T - 1) / T), T, 0, stream>>>(h, dstE, sum0);
        edge_update_wmma<<<NEDGES / 128, T, 0, stream>>>(sum0, srcE, WTupd, a1, h0, h);
    }

    // 3) node aggregation + node layer (WMMA)
    zero_f32<<<(nSum + T - 1) / T, T, 0, stream>>>(sum0, nSum);
    scatter_edge_sum<<<(unsigned)((nScat + T - 1) / T), T, 0, stream>>>(h, dstE, sum0);
    node_out_wmma<<<(NTILES_NODE + 3) / 4, 128, 0, stream>>>(node_feat, sum0, WTnode, b_node, a1, node_h);

    // 4) graph readout
    const int nSol = BGRAPH * NOUTC;
    zero_f32<<<(nSol + T - 1) / T, T, 0, stream>>>(solute, nSol);
    scatter_readout<<<(unsigned)(((long long)NNODES * NOUTC + T - 1) / T), T, 0, stream>>>(node_h, node_graph, solute);

    // 5) MoE head
    build_x<<<(BGRAPH * MOE_IN + T - 1) / T, T, 0, stream>>>(solute, extra, xmoe);
    moe_expert<<<BGRAPH * KEXP, HID, 0, stream>>>(xmoe, We1, be1, We2, be2, We3, be3, a_moe, eo);
    moe_gate<<<BGRAPH, HID, 0, stream>>>(xmoe, Wg1, bg1, Wg2, bg2, Wg3, bg3, a_moe, eo, W_f, b_f, out);
}
